// ChebGraphConv_51513837748778
// MI455X (gfx1250) — matrix operand order repaired and checked by reading, not compiled, not run on hardware
//
#include <hip/hip_runtime.h>
#include <hip/hip_bf16.h>

#define FDIM 256
#define B_SLICE_BYTES 16384   // one K-step of packed B: 32x256 bf16 = 16 KB

typedef __attribute__((ext_vector_type(16))) __bf16 v16bf;
typedef __attribute__((ext_vector_type(8)))  float  v8f;
typedef __attribute__((ext_vector_type(4)))  int    v4i;

typedef __attribute__((address_space(1))) v4i* as1_v4i_ptr;
typedef __attribute__((address_space(3))) v4i* as3_v4i_ptr;

// ---------------------------------------------------------------- zero fill
__global__ void zero_f4_kernel(float4* __restrict__ p, size_t n4) {
  size_t i      = (size_t)blockIdx.x * blockDim.x + threadIdx.x;
  size_t stride = (size_t)gridDim.x * blockDim.x;
  float4 z; z.x = z.y = z.z = z.w = 0.f;
  for (; i < n4; i += stride) p[i] = z;
}

// ---------------------------------------------------------------- SPMM
__global__ void spmm_atomic_kernel(const float* __restrict__ v,
                                   const int*   __restrict__ erow,
                                   const int*   __restrict__ ecol,
                                   const float* __restrict__ ew,
                                   float* out, int n_edges) {
  __shared__ int   s_row[256];
  __shared__ int   s_col[256];
  __shared__ float s_w[256];
  const int f  = threadIdx.x;
  const int e0 = blockIdx.x * 256;
  int cnt = n_edges - e0;
  if (cnt <= 0) return;
  if (cnt > 256) cnt = 256;
  if (f < cnt) {
    s_row[f] = erow[e0 + f];
    s_col[f] = ecol[e0 + f];
    s_w[f]   = ew[e0 + f];
  }
  __syncthreads();
  float acc = 0.f;
  int   cur = s_row[0];
  for (int e = 0; e < cnt; ++e) {
    const int r = s_row[e];
    if (r != cur) {
      atomicAdd(&out[(size_t)cur * FDIM + f], acc);
      acc = 0.f;
      cur = r;
    }
    if (e + 8 < cnt)  // hide L2 gather latency (global_prefetch_b8)
      __builtin_prefetch(&v[(size_t)s_col[e + 8] * FDIM + f], 0, 0);
    acc = fmaf(s_w[e], v[(size_t)s_col[e] * FDIM + f], acc);
  }
  atomicAdd(&out[(size_t)cur * FDIM + f], acc);
}

// ---------------------------------------------------------------- weight prep
// Effective stacked B (768 x 256), bf16, packed in WMMA B-fragment order:
//   fragment (kt,nt), lane l, elem e:  K = kt*32 + (e>>3)*16 + (l>>4)*8 + (e&7),
//                                      N = nt*16 + (l&15)
__global__ void pack_weights_kernel(const float* __restrict__ W,
                                    __bf16* __restrict__ Bp) {
  const int idx = blockIdx.x * 256 + threadIdx.x;   // < 196608
  const int e   = idx & 15;
  const int l   = (idx >> 4) & 31;
  const int nt  = (idx >> 9) & 15;
  const int kt  = idx >> 13;
  const int k   = kt * 32 + (e >> 3) * 16 + (l >> 4) * 8 + (e & 7);      // 0..767
  const int n   = nt * 16 + (l & 15);               // 0..255
  float val;
  if (k < 256)
    val = W[(size_t)k * 256 + n] - W[(size_t)(512 + k) * 256 + n]; // W0 - W2
  else if (k < 512)
    val = W[(size_t)k * 256 + n];                                   // W1
  else
    val = 2.f * W[(size_t)k * 256 + n];                             // 2*W2
  Bp[idx] = (__bf16)val;
}

// ---------------------------------------------------------------- async stage
__device__ __forceinline__ void wait_async0() {
#if __has_builtin(__builtin_amdgcn_s_wait_asynccnt)
  __builtin_amdgcn_s_wait_asynccnt(0);
#else
  asm volatile("s_wait_asynccnt 0x0" ::: "memory");
#endif
}

// Cooperatively stage one 16 KB B K-slice into LDS: 128 threads x 8 x b128.
__device__ __forceinline__ void stage_b(const __bf16* Bp, int kt, char* dst,
                                        int tid) {
  const char* g = (const char*)Bp + (size_t)kt * B_SLICE_BYTES + (size_t)tid * 16;
  char* l = dst + tid * 16;
#pragma unroll
  for (int i = 0; i < 8; ++i) {
#if __has_builtin(__builtin_amdgcn_global_load_async_to_lds_b128)
    __builtin_amdgcn_global_load_async_to_lds_b128(
        (as1_v4i_ptr)(g + i * 2048),
        (as3_v4i_ptr)(l + i * 2048),
        0, 0);
#else
    const unsigned lds_off =
        (unsigned)(size_t)(__attribute__((address_space(3))) char*)(l + i * 2048);
    const unsigned long long gaddr = (unsigned long long)(size_t)(g + i * 2048);
    asm volatile("global_load_async_to_lds_b128 %0, %1, off"
                 :: "v"(lds_off), "v"(gaddr) : "memory");
#endif
  }
}

// ---------------------------------------------------------------- GEMM
__device__ __forceinline__ v16bf cvt_f32x16_bf16(const float4& a0, const float4& a1,
                                                 const float4& a2, const float4& a3) {
  v16bf r;
  r[0]  = (__bf16)a0.x; r[1]  = (__bf16)a0.y; r[2]  = (__bf16)a0.z; r[3]  = (__bf16)a0.w;
  r[4]  = (__bf16)a1.x; r[5]  = (__bf16)a1.y; r[6]  = (__bf16)a1.z; r[7]  = (__bf16)a1.w;
  r[8]  = (__bf16)a2.x; r[9]  = (__bf16)a2.y; r[10] = (__bf16)a2.z; r[11] = (__bf16)a2.w;
  r[12] = (__bf16)a3.x; r[13] = (__bf16)a3.y; r[14] = (__bf16)a3.z; r[15] = (__bf16)a3.w;
  return r;
}

// out[M,256] = [x | T1 | S] (M x 768) @ Bp (768 x 256, bf16) + bias.
// One wave per 16-row M-tile; B double-buffered in LDS via async-to-LDS,
// staged once per block and broadcast to all 4 waves through ds_load.
// A sched_group_barrier pipeline keeps one B fragment (2 ds_load_b128) in
// flight behind every WMMA so the matrix pipe is not gated on dscnt==0.
// s aliases out (S staged in d_out): each wave reads only its own 16 rows of
// s inside the K-loop, then writes those same rows; no __restrict__ on either.
__global__ void __launch_bounds__(128)
cheb_gemm_kernel(const float* __restrict__ x,
                 const float* __restrict__ t1,
                 const float* s,
                 const __bf16* __restrict__ Bp,
                 const float* __restrict__ bias,
                 float* out, int n, int mtiles) {
  __shared__ __align__(16) char lds_b[2][B_SLICE_BYTES];

  const int tid   = threadIdx.x;
  const int wave  = tid >> 5;
  const int lane  = tid & 31;
  int  mtile      = blockIdx.x * 4 + wave;
  const bool active = (mtile < mtiles);   // no early return: barriers below
  if (!active) mtile = mtiles - 1;        // clamp; stores masked by `active`
  const int m0    = mtile * 16;
  const int khalf = lane >> 4;
  int mrow = m0 + (lane & 15);
  if (mrow >= n) mrow = n - 1;

  const v8f vzero = {};
  v8f c[16];
#pragma unroll
  for (int i = 0; i < 16; ++i) c[i] = vzero;

  stage_b(Bp, 0, lds_b[0], tid);          // prime the LDS pipeline

  // Prime the A pipeline (kt = 0 reads from x).
  const size_t arow_off = (size_t)mrow * FDIM + khalf * 8;
  const float* ap = x + arow_off;
  float4 a0 = *(const float4*)(ap + 0);
  float4 a1 = *(const float4*)(ap + 4);
  float4 a2 = *(const float4*)(ap + 16);
  float4 a3 = *(const float4*)(ap + 20);

  for (int kt = 0; kt < 24; ++kt) {
    wait_async0();                        // this wave's staged portion landed
    __syncthreads();                      // slice kt visible; buf (kt+1)&1 free
    if (kt + 1 < 24) stage_b(Bp, kt + 1, lds_b[(kt + 1) & 1], tid);

    // 16-bit A layout: lanes 0-15 hold K 0-7 & 16-23, lanes 16-31 hold
    // K 8-15 & 24-31 (fragment elements 0..7 then 8..15).
    const v16bf a = cvt_f32x16_bf16(a0, a1, a2, a3);

    if (kt + 1 < 24) {                    // prefetch next K-step of A
      const int ktn = kt + 1;
      const float* srcn = (ktn < 8) ? x : (ktn < 16) ? t1 : s;
      const float* apn = srcn + arow_off + (ktn & 7) * 32;
      a0 = *(const float4*)(apn + 0);
      a1 = *(const float4*)(apn + 4);
      a2 = *(const float4*)(apn + 16);
      a3 = *(const float4*)(apn + 20);
    }

    const char* bbuf = lds_b[kt & 1];
#pragma unroll
    for (int nt = 0; nt < 16; ++nt) {
      const v16bf b = *(const v16bf*)(bbuf + nt * 1024 + lane * 32);
      c[nt] = __builtin_amdgcn_wmma_f32_16x16x32_bf16(
          false, a, false, b, (short)0, c[nt], false, false);
    }

#if __has_builtin(__builtin_amdgcn_sched_group_barrier)
    // Pipeline request for this region: 32 DS reads + 16 WMMAs.
    //   [4 ds reads] then 14 x [1 wmma, 2 ds reads] then [2 wmma]
    // => every WMMA has the *next* fragment's loads already in flight.
    __builtin_amdgcn_sched_group_barrier(0x100, 4, 0);   // DS read
#pragma unroll
    for (int i = 0; i < 14; ++i) {
      __builtin_amdgcn_sched_group_barrier(0x008, 1, 0); // MFMA/WMMA
      __builtin_amdgcn_sched_group_barrier(0x100, 2, 0); // DS read
    }
    __builtin_amdgcn_sched_group_barrier(0x008, 2, 0);   // final WMMAs
#endif
  }

  // Epilogue: C layout lane l, vgpr v -> M = m0 + (l>>4)*8 + v, N = (l&15).
  const int nlane = lane & 15;
  const int mbase = m0 + khalf * 8;
#pragma unroll
  for (int nt = 0; nt < 16; ++nt) {
    const float bv = bias[nt * 16 + nlane];
    float* op = out + (size_t)mbase * FDIM + nt * 16 + nlane;
#pragma unroll
    for (int v = 0; v < 8; ++v) {
      if (active && (mbase + v < n)) op[(size_t)v * FDIM] = c[nt][v] + bv;
    }
  }
}

// ---------------------------------------------------------------- launch
extern "C" void kernel_launch(void* const* d_in, const int* in_sizes, int n_in,
                              void* d_out, int out_size, void* d_ws, size_t ws_size,
                              hipStream_t stream) {
  const float* x    = (const float*)d_in[0];
  const int*   erow = (const int*)d_in[1];
  const int*   ecol = (const int*)d_in[2];
  const float* ew   = (const float*)d_in[3];
  const float* W    = (const float*)d_in[4];
  const float* bias = (const float*)d_in[5];
  const int n_nodes = in_sizes[0] / FDIM;
  const int n_edges = in_sizes[1];

  // Workspace: T1 (n*256 f32) then packed bf16 weights (768*256).
  float*  T1 = (float*)d_ws;
  size_t  t1_bytes = (size_t)n_nodes * FDIM * sizeof(float);
  __bf16* Bp = (__bf16*)((char*)d_ws + t1_bytes);
  float*  S  = (float*)d_out;   // S = spmm(T1) staged in d_out
  float*  out = (float*)d_out;

  const size_t n4 = (size_t)n_nodes * FDIM / 4;
  const int eblocks = (n_edges + 255) / 256;

  zero_f4_kernel<<<2048, 256, 0, stream>>>((float4*)T1, n4);
  spmm_atomic_kernel<<<eblocks, 256, 0, stream>>>(x, erow, ecol, ew, T1, n_edges);
  zero_f4_kernel<<<2048, 256, 0, stream>>>((float4*)S, n4);
  spmm_atomic_kernel<<<eblocks, 256, 0, stream>>>(T1, erow, ecol, ew, S, n_edges);
  pack_weights_kernel<<<(3 * FDIM * FDIM) / 256, 256, 0, stream>>>(W, Bp);

  const int mtiles  = (n_nodes + 15) / 16;
  const int gblocks = (mtiles + 3) / 4;
  cheb_gemm_kernel<<<gblocks, 128, 0, stream>>>(x, T1, S, Bp, bias, out,
                                                n_nodes, mtiles);
}
